// BiLSTM_CRF_44865228374181
// MI455X (gfx1250) — matrix operand order repaired and checked by reading, not compiled, not run on hardware
//
#include <hip/hip_runtime.h>
#include <hip/hip_bf16.h>
#include <math.h>

// ---------------------------------------------------------------------------
// BiLSTM-CRF loss for MI455X (gfx1250, wave32, WMMA + TDM).
//   phase 0: embedding gather + f16 conversion of x / Wih / Whh
//   phase 1: batched input projection  G = x @ Wih^T + b   (v_wmma f16->f32)
//   phase 2: recurrent LSTM, 1 persistent workgroup per direction,
//            Whh resident in LDS via TENSOR_LOAD_TO_LDS, per-step gate block
//            double-buffered into LDS by the Tensor Data Mover while the
//            waves run the (32x128 @ 128x512) gate GEMM with v_wmma.
//   phase 3: feats = [hf,hb] @ W_tag^T + b_tag   (tiny, scalar FMA)
//   phase 4: CRF forward recursion + gold score, 1 lane per batch element
// ---------------------------------------------------------------------------

typedef __attribute__((ext_vector_type(16))) _Float16 v16h;
typedef __attribute__((ext_vector_type(8)))  float    v8f;
typedef __attribute__((ext_vector_type(4)))  unsigned int v4u;
typedef __attribute__((ext_vector_type(8)))  int      v8i;
typedef __attribute__((ext_vector_type(4)))  int      v4i;

#define EMB      256
#define HID      256
#define HH       128   // H = HID/2
#define G4H      512   // 4*H
#define LEN      512
#define BATCH    32
#define NT       5
#define START_TAG 3
#define STOP_TAG  4

// ---- WMMA fragment loaders (16x16x32 f16, layouts per CDNA5 ISA 7.12.2) ----

// A matrix 16x32 (MxK), row-major source with leading dim `ld`.
// lanes 0-15: V0..3 = K 0..7, V4..7 = K 16..23 ; lanes 16-31: K 8..15 / 24..31
__device__ __forceinline__ v16h frag_a(const _Float16* base, int ld,
                                       int m0, int k0, int lane) {
  int row = m0 + (lane & 15);
  int hi  = (lane >> 4) & 1;
  const _Float16* p0 = base + row * ld + k0 + hi * 8;
  const _Float16* p1 = p0 + 16;
  v16h a;
#pragma unroll
  for (int e = 0; e < 8; ++e) { a[e] = p0[e]; a[8 + e] = p1[e]; }
  return a;
}

// B matrix 32x16 (KxN) sourced from W stored (N x K) row-major: B[k][n]=W[n][k].
// lanes 0-15: K 0..7 / 16..23, lanes 16-31: K 8..15 / 24..31; column = lane&15.
__device__ __forceinline__ v16h frag_b(const _Float16* base, int ld,
                                       int n0, int k0, int lane) {
  int col = n0 + (lane & 15);
  int hi  = (lane >> 4) & 1;
  const _Float16* p0 = base + col * ld + k0 + hi * 8;
  const _Float16* p1 = p0 + 16;
  v16h b;
#pragma unroll
  for (int e = 0; e < 8; ++e) { b[e] = p0[e]; b[8 + e] = p1[e]; }
  return b;
}

__device__ __forceinline__ float sigf(float x) {
  return 1.0f / (1.0f + __expf(-x));
}

// ---- Tensor Data Mover: 1-D contiguous global -> LDS DMA ------------------
// D# per CDNA5 ISA 8.3/8.4: group0 = {count=1, lds_addr, global_addr, type=2},
// group1 = {data_size=4B, tensor_dim0/tile_dim0/stride in dword units}.
// 6-arg builtin form (clang-23 / therock headers): (v4u, v8i, v4i, v4i, v8i, i32)
__device__ __forceinline__ void tdm_load_1d(unsigned lds_off,
                                            const void* gptr,
                                            unsigned ndwords) {
  unsigned long long ga = (unsigned long long)(uintptr_t)gptr;
  v4u g0;
  g0[0] = 1u;                                         // count=1, user D#
  g0[1] = lds_off;                                    // lds_addr (bytes)
  g0[2] = (unsigned)(ga & 0xFFFFFFFFu);               // global_addr[31:0]
  g0[3] = (unsigned)((ga >> 32) & 0x01FFFFFFu)        // global_addr[56:32]
        | (2u << 30);                                 // type = 2 ("image")
  v8i g1 = {0, 0, 0, 0, 0, 0, 0, 0};
  g1[0] = (int)(2u << 16);                            // data_size = 4 bytes
  g1[1] = (int)((ndwords & 0xFFFFu) << 16);           // tensor_dim0[15:0]
  g1[2] = (int)((ndwords >> 16) & 0xFFFFu);           // tensor_dim0[31:16]
  g1[3] = (int)((ndwords & 0xFFFFu) << 16);           // tile_dim0
  g1[5] = (int)ndwords;                               // tensor_dim0_stride
  v4i z4 = {0, 0, 0, 0};
  v8i z8 = {0, 0, 0, 0, 0, 0, 0, 0};
  __builtin_amdgcn_tensor_load_to_lds(g0, g1, z4, z4, z8, 0);
}

// ---------------------------------------------------------------------------
// Phase 0: gather embeddings into x16 (m = t*B + b, row-major EMB) and convert
// Wih_f|Wih_b -> wih16 (2 dirs concat), Whh_f|Whh_b -> whh16.
// ---------------------------------------------------------------------------
__global__ void __launch_bounds__(256)
prep_kernel(const int* __restrict__ sentence, const float* __restrict__ embed,
            const float* __restrict__ Wih_f, const float* __restrict__ Wih_b,
            const float* __restrict__ Whh_f, const float* __restrict__ Whh_b,
            _Float16* __restrict__ x16, _Float16* __restrict__ wih16,
            _Float16* __restrict__ whh16) {
  const int NX   = LEN * BATCH * EMB;   // 4,194,304
  const int NWIH = G4H * EMB;           // 131,072 per dir
  const int NWHH = G4H * HH;            // 65,536 per dir
  const int total = NX + 2 * NWIH + 2 * NWHH;
  for (int idx = blockIdx.x * blockDim.x + threadIdx.x; idx < total;
       idx += gridDim.x * blockDim.x) {
    if (idx < NX) {
      int m = idx >> 8;          // / EMB
      int e = idx & (EMB - 1);
      int t = m >> 5;            // / BATCH
      int b = m & (BATCH - 1);
      int tok = sentence[b * LEN + t];
      x16[idx] = (_Float16)embed[(size_t)tok * EMB + e];
    } else if (idx < NX + 2 * NWIH) {
      int j = idx - NX;
      if (j < NWIH) wih16[j] = (_Float16)Wih_f[j];
      else          wih16[j] = (_Float16)Wih_b[j - NWIH];
    } else {
      int j = idx - NX - 2 * NWIH;
      if (j < NWHH) whh16[j] = (_Float16)Whh_f[j];
      else          whh16[j] = (_Float16)Whh_b[j - NWHH];
    }
  }
}

// ---------------------------------------------------------------------------
// Phase 1: G[dir] = x16 @ Wih[dir]^T + b[dir]    (M=L*B=16384, N=512, K=256)
// 8 waves per block, one 16x16 tile per wave, 8 chained WMMAs over K.
// Output stored f16 (gate pre-activations re-read every LSTM step).
// ---------------------------------------------------------------------------
__global__ void __launch_bounds__(256)
gemm_in_kernel(const _Float16* __restrict__ x16, const _Float16* __restrict__ wih16,
               const float* __restrict__ b_f, const float* __restrict__ b_b,
               _Float16* __restrict__ G) {
  const int tilesPerDir = (LEN * BATCH / 16) * (G4H / 16); // 1024*32 = 32768
  int gw   = blockIdx.x * (blockDim.x >> 5) + (threadIdx.x >> 5);
  int lane = threadIdx.x & 31;
  int dir  = gw / tilesPerDir;
  int rem  = gw - dir * tilesPerDir;
  int mTile = rem >> 5;
  int nTile = rem & 31;
  const _Float16* W    = wih16 + (size_t)dir * G4H * EMB;
  const float*    bias = dir ? b_b : b_f;
  _Float16*       Gd   = G + (size_t)dir * (size_t)LEN * BATCH * G4H;

  int m0 = mTile * 16, n0 = nTile * 16;
  int col = n0 + (lane & 15);
  int hi  = (lane >> 4) & 1;
  float bv = bias[col];
  v8f c;
#pragma unroll
  for (int r = 0; r < 8; ++r) c[r] = bv;

#pragma unroll
  for (int kk = 0; kk < EMB / 32; ++kk) {
    v16h a = frag_a(x16, EMB, m0, kk * 32, lane);
    v16h b = frag_b(W,   EMB, n0, kk * 32, lane);
    c = __builtin_amdgcn_wmma_f32_16x16x32_f16(false, a, false, b,
                                               (short)0, c, false, false);
  }
#pragma unroll
  for (int r = 0; r < 8; ++r) {
    int m = m0 + r + hi * 8;
    Gd[(size_t)m * G4H + col] = (_Float16)c[r];
  }
}

// ---------------------------------------------------------------------------
// Phase 2: recurrent LSTM. blockIdx.x = direction. 256 threads = 8 waves.
// LDS: Whh f16 (128K, TDM-loaded) + h f16 (8K) + c f32 (16K) + gates f32 (64K)
//      + G ping-pong (2x32K, TDM double-buffered)  = 280KB < 320KB WGP LDS.
// Per step: wave 0 issues TENSOR_LOAD_TO_LDS for step s+1's gate block, then
// s_wait_tensorcnt<=1 guarantees step s's buffer is resident (TDM ops from
// one wave complete in order); barrier publishes it; 8 waves run 64 WMMA
// tiles; elementwise sigmoid/tanh cell update closes the step.
// ---------------------------------------------------------------------------
__global__ void __launch_bounds__(256)
lstm_kernel(const _Float16* __restrict__ whh16, const _Float16* __restrict__ G,
            const float* __restrict__ h0, const float* __restrict__ c0,
            float* __restrict__ hout) {
  extern __shared__ char smem[];
  _Float16* whh  = (_Float16*)smem;              // [G4H*HH]      128KB
  _Float16* hsh  = whh + G4H * HH;               // [BATCH*HH]      8KB
  float*    csh  = (float*)(hsh + BATCH * HH);   // [BATCH*HH]     16KB
  float*    gsh  = csh + BATCH * HH;             // [BATCH*G4H]    64KB
  _Float16* gbuf = (_Float16*)(gsh + BATCH * G4H); // [2][BATCH*G4H] 64KB

  const int dir  = blockIdx.x;
  const int tid  = threadIdx.x;
  const int lane = tid & 31;
  const int wave = tid >> 5;

  const _Float16* Gd = G + (size_t)dir * (size_t)LEN * BATCH * G4H;
  float* hd = hout + (size_t)dir * (size_t)LEN * BATCH * HH;
  const int tau0 = dir ? (LEN - 1) : 0;

  if (wave == 0) {
    // TDM: Whh (128KB = 32768 dwords) and first gate block (32KB = 8192 dw)
    tdm_load_1d((unsigned)(uintptr_t)whh,
                whh16 + (size_t)dir * G4H * HH, (G4H * HH) / 2);
    tdm_load_1d((unsigned)(uintptr_t)gbuf,
                Gd + (size_t)tau0 * BATCH * G4H, (BATCH * G4H) / 2);
  }
  for (int i = tid; i < BATCH * HH; i += blockDim.x) {
    hsh[i] = (_Float16)h0[(size_t)dir * BATCH * HH + i];
    csh[i] = c0[(size_t)dir * BATCH * HH + i];
  }

  for (int s = 0; s < LEN; ++s) {
    int tau = dir ? (LEN - 1 - s) : s;
    if (wave == 0) {
      if (s + 1 < LEN) {  // prefetch next step's gate block into other buffer
        int tn = dir ? (LEN - 2 - s) : (s + 1);
        tdm_load_1d((unsigned)(uintptr_t)(gbuf + ((s + 1) & 1) * BATCH * G4H),
                    Gd + (size_t)tn * BATCH * G4H, (BATCH * G4H) / 2);
        __builtin_amdgcn_s_wait_tensorcnt(1);  // step-s buffer resident
      } else {
        __builtin_amdgcn_s_wait_tensorcnt(0);
      }
    }
    __syncthreads();  // publish TDM-written LDS to all waves

    const _Float16* Gs = gbuf + (s & 1) * BATCH * G4H;

    // gate GEMM: g = G(tau) + h @ Whh^T   (M=32, N=512, K=128)
    for (int i = 0; i < 8; ++i) {
      int tt = wave * 8 + i;
      int mTile = tt >> 5;
      int nTile = tt & 31;
      int m0 = mTile * 16, n0 = nTile * 16;
      int col = n0 + (lane & 15);
      int hi  = (lane >> 4) & 1;
      v8f c;
#pragma unroll
      for (int r = 0; r < 8; ++r)
        c[r] = (float)Gs[(m0 + r + hi * 8) * G4H + col];
#pragma unroll
      for (int kk = 0; kk < HH / 32; ++kk) {
        v16h a = frag_a(hsh, HH, m0, kk * 32, lane);
        v16h b = frag_b(whh, HH, n0, kk * 32, lane);
        c = __builtin_amdgcn_wmma_f32_16x16x32_f16(false, a, false, b,
                                                   (short)0, c, false, false);
      }
#pragma unroll
      for (int r = 0; r < 8; ++r)
        gsh[(m0 + r + hi * 8) * G4H + col] = c[r];
    }
    __syncthreads();

    // elementwise cell update: gate order is i, f, g, o (jnp.split of 4H)
    for (int idx = tid; idx < BATCH * HH; idx += blockDim.x) {
      int b = idx >> 7;
      int j = idx & (HH - 1);
      float ig = gsh[b * G4H + j];
      float fg = gsh[b * G4H + HH + j];
      float gg = gsh[b * G4H + 2 * HH + j];
      float og = gsh[b * G4H + 3 * HH + j];
      float cc = sigf(fg) * csh[idx] + sigf(ig) * tanhf(gg);
      float hh = sigf(og) * tanhf(cc);
      csh[idx] = cc;
      hsh[idx] = (_Float16)hh;
      hd[((size_t)tau * BATCH + b) * HH + j] = hh;
    }
    __syncthreads();
  }
}

// ---------------------------------------------------------------------------
// Phase 3: feats[t,b,k] = hf[t,b,:].W_tag[k,:128] + hb[t,b,:].W_tag[k,128:] + b_tag[k]
// ---------------------------------------------------------------------------
__global__ void __launch_bounds__(256)
feats_kernel(const float* __restrict__ hout, const float* __restrict__ W_tag,
             const float* __restrict__ b_tag, float* __restrict__ feats) {
  int idx = blockIdx.x * blockDim.x + threadIdx.x;
  if (idx >= LEN * BATCH * NT) return;
  int k  = idx % NT;
  int mb = idx / NT;  // t*B + b
  const float* hf = hout + (size_t)mb * HH;
  const float* hb = hout + (size_t)LEN * BATCH * HH + (size_t)mb * HH;
  const float* w  = W_tag + k * HID;
  float acc = b_tag[k];
#pragma unroll 8
  for (int j = 0; j < HH; ++j) acc += hf[j] * w[j] + hb[j] * w[HH + j];
  feats[idx] = acc;
}

// ---------------------------------------------------------------------------
// Phase 4: CRF forward score + gold score. One lane per batch element (B=32
// == one wave32). T=5 recursion kept in registers.
// ---------------------------------------------------------------------------
__global__ void __launch_bounds__(32)
crf_kernel(const float* __restrict__ feats, const float* __restrict__ trans,
           const int* __restrict__ tags, const int* __restrict__ mask,
           float* __restrict__ out) {
  __shared__ float red[BATCH];
  int b = threadIdx.x;

  float tr[NT * NT];
#pragma unroll
  for (int i = 0; i < NT * NT; ++i) tr[i] = trans[i];

  float alpha[NT], saved[NT];
#pragma unroll
  for (int k = 0; k < NT; ++k) alpha[k] = -10000.0f;
  alpha[START_TAG] = 0.0f;
#pragma unroll
  for (int k = 0; k < NT; ++k) saved[k] = alpha[k];

  int mb = mask[b];
  for (int t = 1; t <= LEN - 2; ++t) {
    const float* ft = feats + ((size_t)t * BATCH + b) * NT;
    float na[NT];
#pragma unroll
    for (int k = 0; k < NT; ++k) {
      float mx = -3.0e38f;
#pragma unroll
      for (int j = 0; j < NT; ++j) mx = fmaxf(mx, alpha[j] + tr[k * NT + j]);
      float sum = 0.0f;
#pragma unroll
      for (int j = 0; j < NT; ++j) sum += __expf(alpha[j] + tr[k * NT + j] - mx);
      na[k] = ft[k] + mx + __logf(sum);
    }
#pragma unroll
    for (int k = 0; k < NT; ++k) alpha[k] = na[k];
    if (t == mb) {
#pragma unroll
      for (int k = 0; k < NT; ++k) saved[k] = alpha[k];
    }
  }

  float mx = -3.0e38f;
#pragma unroll
  for (int k = 0; k < NT; ++k)
    mx = fmaxf(mx, saved[k] + tr[STOP_TAG * NT + k]);
  float sum = 0.0f;
#pragma unroll
  for (int k = 0; k < NT; ++k)
    sum += __expf(saved[k] + tr[STOP_TAG * NT + k] - mx);
  float fwd = mx + __logf(sum);

  float gold = 0.0f;
  for (int i = 1; i <= mb; ++i) {
    int tp = tags[b * LEN + i - 1];
    int tn = tags[b * LEN + i];
    gold += tr[tn * NT + tp] + feats[((size_t)i * BATCH + b) * NT + tn];
  }
  gold += tr[STOP_TAG * NT + tags[b * LEN + mb]];

  red[b] = fwd - gold;
  __syncthreads();
  if (b == 0) {
    float s = 0.0f;
    for (int i = 0; i < BATCH; ++i) s += red[i];
    out[0] = s / (float)BATCH;
  }
}

// ---------------------------------------------------------------------------
extern "C" void kernel_launch(void* const* d_in, const int* in_sizes, int n_in,
                              void* d_out, int out_size, void* d_ws, size_t ws_size,
                              hipStream_t stream) {
  const int*   sentence = (const int*)d_in[0];
  const int*   tags     = (const int*)d_in[1];
  const int*   mask     = (const int*)d_in[2];
  const float* embed    = (const float*)d_in[3];
  const float* Wih_f    = (const float*)d_in[4];
  const float* Whh_f    = (const float*)d_in[5];
  const float* b_f      = (const float*)d_in[6];
  const float* Wih_b    = (const float*)d_in[7];
  const float* Whh_b    = (const float*)d_in[8];
  const float* b_b      = (const float*)d_in[9];
  const float* W_tag    = (const float*)d_in[10];
  const float* b_tag    = (const float*)d_in[11];
  const float* trans    = (const float*)d_in[12];
  const float* h0       = (const float*)d_in[13];
  const float* c0       = (const float*)d_in[14];

  char* ws = (char*)d_ws;
  // workspace layout (bytes, 256-aligned):
  _Float16* x16   = (_Float16*)(ws + 0);          //  8,388,608  x f16 (L*B, EMB)
  _Float16* wih16 = (_Float16*)(ws + 8388608);    //    524,288  Wih f16, 2 dirs
  _Float16* whh16 = (_Float16*)(ws + 8912896);    //    262,144  Whh f16, 2 dirs
  _Float16* G     = (_Float16*)(ws + 9175040);    // 33,554,432  gate preacts, 2 dirs
  float*    hout  = (float*)   (ws + 42729472);   // 16,777,216  hf|hb f32
  float*    feats = (float*)   (ws + 59506688);   //    327,680  (L, B, T) f32

  prep_kernel<<<2048, 256, 0, stream>>>(sentence, embed, Wih_f, Wih_b,
                                        Whh_f, Whh_b, x16, wih16, whh16);

  // 2 dirs * 1024 M-tiles * 32 N-tiles = 65536 wave-tiles / 8 waves per block
  gemm_in_kernel<<<8192, 256, 0, stream>>>(x16, wih16, b_f, b_b, G);

  size_t smem = (size_t)G4H * HH * 2      // Whh f16    131072
              + (size_t)BATCH * HH * 2    // h   f16      8192
              + (size_t)BATCH * HH * 4    // c   f32     16384
              + (size_t)BATCH * G4H * 4   // gates f32   65536
              + 2u * BATCH * G4H * 2;     // G ping-pong 65536  => 286720 < 320KB
  lstm_kernel<<<2, 256, smem, stream>>>(whh16, G, h0, c0, hout);

  feats_kernel<<<(LEN * BATCH * NT + 255) / 256, 256, 0, stream>>>(
      hout, W_tag, b_tag, feats);

  crf_kernel<<<1, 32, 0, stream>>>(feats, trans, tags, mask, (float*)d_out);
}
